// TT_Net_54254026883466
// MI455X (gfx1250) — matrix operand order repaired and checked by reading, not compiled, not run on hardware
//
#include <hip/hip_runtime.h>
#include <hip/hip_bf16.h>

typedef __attribute__((ext_vector_type(16))) _Float16 v16h;
typedef __attribute__((ext_vector_type(8)))  float    v8f;

// ---------------------------------------------------------------------------
// One TT-core contraction as GEMM:  C[M,N] = A[M,K] x G[K,N]
//   M = 512*RO (RO = rest*out), K = rp*m <= 32, N = RN*NM <= 32.
//   A addr : (b*K + k)*RO + rem            (row i = b*RO + rem)
//   G addr : (k*NM + nn)*RN + q            (col j = q*NM + nn)
//   C addr : ((b*RN + q)*RO + rem)*NM + nn
// Fully compile-time shapes: every load/store offset is an immediate, no
// runtime division, no divergent guards. One wave handles one N-tile column
// and TM consecutive M-tiles, reusing the B fragment across TM WMMAs.
// Intermediates are stored as f16 (halves HBM traffic; WMMA input precision
// anyway); only the first step reads the f32 network input.
// ---------------------------------------------------------------------------
template<typename TA, int K, int RO, int NM, int RN, int TM>
__global__ __launch_bounds__(32)
void tt_step(const TA* __restrict__ A, const float* __restrict__ G,
             _Float16* __restrict__ O) {
  constexpr int N  = RN * NM;
  constexpr int NT = (N + 15) / 16;
  constexpr int GS = NM * RN;             // G stride per k
  static_assert(K <= 32 && N <= 32, "");
  static_assert(RO % 16 == 0, "16 rows of a tile stay in one batch block");
  static_assert((32 * RO) % TM == 0, "tile loop has no remainder");

  const int wave = blockIdx.x;            // blockDim.x == 32: scalar wave id
  const int nt   = (NT == 1) ? 0 : (wave % NT);
  const int mc   = (NT == 1) ? wave : (wave / NT);
  const int lane = threadIdx.x & 31;
  const int half = lane >> 4;             // wave32 lane halves
  const int l15  = lane & 15;

  // ---- column decomposition (compile-time shifts; clamp when N < 16) ----
  const int  j   = nt * 16 + l15;
  const bool jok = (N >= 16) || (j < N);
  const int  jc  = (N >= 16) ? j : (jok ? j : 0);
  const int  q   = jc / NM;               // NM is a power of two
  const int  nn  = jc - q * NM;

  // ---- B fragment (loaded once, reused for TM WMMAs) --------------------
  // k order chosen here (contiguous, same as A): lanes 0-15 hold K=0..15, lanes 16-31 hold K=16..31.
  v16h bf;
  {
    const float* Gb  = G + (size_t)nn * RN + q;
    const float* Gbh = Gb + (size_t)half * 16 * GS;   // fold half into base
#pragma unroll
    for (int e = 0; e < 16; ++e) {
      float v;
      if (16 + e < K) {                   // valid for both halves
        v = Gbh[(size_t)e * GS];
      } else if (e >= K) {                // invalid for both halves
        v = 0.0f;
      } else {                            // valid only for half==0
        float t0 = Gb[(size_t)e * GS];    // half==0 address, always in-bounds
        v = half ? 0.0f : t0;
      }
      if (N < 16) v = jok ? v : 0.0f;
      bf[e] = (_Float16)v;
    }
  }

#pragma unroll
  for (int t = 0; t < TM; ++t) {
    const int mt   = mc * TM + t;
    const int i0   = mt * 16;             // 16 | RO -> single batch block
    const int b    = i0 / RO;             // const division -> mulhi/shift
    const int rem0 = i0 - b * RO;

    // ---- A fragment ----
    // k order chosen here (contiguous, same as B): lanes 0-15: K=0-15 ; lanes 16-31: K=16-31.
    const TA* Ab  = A + (size_t)b * (K * RO) + rem0 + l15;
    const TA* Abh = Ab + (size_t)half * 16 * RO;      // fold half into base
    v16h a;
#pragma unroll
    for (int run = 0; run < 2; ++run) {
#pragma unroll
      for (int e = 0; e < 8; ++e) {
        const int kmin = run * 8 + e;     // k for half==0
        _Float16 v;
        if (kmin + 16 < K) {              // valid for both halves
          v = (_Float16)Abh[(size_t)kmin * RO];
        } else if (kmin >= K) {           // invalid for both halves
          v = (_Float16)0.0f;
        } else {                          // valid only for half==0
          _Float16 t0 = (_Float16)Ab[(size_t)kmin * RO];  // in-bounds
          v = half ? (_Float16)0.0f : t0;
        }
        a[run * 8 + e] = v;
      }
    }

    v8f c = {};
    c = __builtin_amdgcn_wmma_f32_16x16x32_f16(false, a, false, bf,
                                               (short)0, c, false, false);

    // ---- store: C/D layout: VGPR v -> row = 8*half + v, col = l15 ----
    if (N >= 16 || jok) {
      _Float16* Ob =
          O + ((size_t)(b * RN + q) * RO + rem0 + half * 8) * NM + nn;
#pragma unroll
      for (int v = 0; v < 8; ++v)
        Ob[(size_t)v * NM] = (_Float16)c[v];
    }
  }
}

typedef void (*StepFn)(const void*, const float*, void*, hipStream_t);

template<typename TA, int K, int RO, int NM, int RN>
static void launch_step(const void* A, const float* G, void* O,
                        hipStream_t s) {
  constexpr int TM = 8;
  constexpr int MT = 32 * RO;             // (512*RO)/16
  constexpr int NT = (RN * NM + 15) / 16;
  constexpr int blocks = (MT / TM) * NT;  // one wave per block
  tt_step<TA, K, RO, NM, RN, TM><<<blocks, 32, 0, s>>>(
      (const TA*)A, G, (_Float16*)O);
}

// ---------------------------------------------------------------------------
// Fused bias + training-mode BatchNorm (biased var, f32 stats) + ReLU +
// dropout mask. f16 in / f16 out. One thread per feature column; two
// coalesced passes over the 512 rows.
// ---------------------------------------------------------------------------
__global__ void bn_relu_mask_kernel(const _Float16* __restrict__ t,
                                    const float* __restrict__ bias,
                                    const float* __restrict__ gamma,
                                    const float* __restrict__ beta,
                                    const float* __restrict__ mask,
                                    _Float16* __restrict__ h, int B, int N) {
  const int col = blockIdx.x * blockDim.x + threadIdx.x;
  if (col >= N) return;
  const float bi = bias[col];
  float s = 0.0f, s2 = 0.0f;
  for (int r = 0; r < B; ++r) {
    float v = (float)t[(size_t)r * N + col] + bi;
    s += v; s2 += v * v;
  }
  const float invB = 1.0f / (float)B;
  const float mu   = s * invB;
  const float var  = s2 * invB - mu * mu;
  const float sc   = gamma[col] * rsqrtf(var + 1e-5f);
  const float be   = beta[col];
  for (int r = 0; r < B; ++r) {
    float v = (float)t[(size_t)r * N + col] + bi;
    float y = sc * (v - mu) + be;
    y = y > 0.0f ? y : 0.0f;
    h[(size_t)r * N + col] = (_Float16)(y * mask[(size_t)r * N + col]);
  }
}

// logits[b,j] = h[b,:] . w5[j,:] + bias5[j]   (512x10, K=1024; tiny)
__global__ void dense10_kernel(const _Float16* __restrict__ h,
                               const float* __restrict__ w,
                               const float* __restrict__ bias,
                               float* __restrict__ logits, int B) {
  const int t = blockIdx.x * blockDim.x + threadIdx.x;
  if (t >= B * 10) return;
  const int b = t / 10, j = t - b * 10;
  const _Float16* hr = h + (size_t)b * 1024;
  const float*    wr = w + (size_t)j * 1024;
  float s = bias[j];
  for (int k = 0; k < 1024; ++k) s += (float)hr[k] * wr[k];
  logits[t] = s;
}

__global__ void bn5_kernel(float* __restrict__ logits,
                           const float* __restrict__ gamma,
                           const float* __restrict__ beta, int B) {
  const int col = threadIdx.x;
  if (col >= 10) return;
  float s = 0.0f, s2 = 0.0f;
  for (int r = 0; r < B; ++r) { float v = logits[r * 10 + col]; s += v; s2 += v * v; }
  const float invB = 1.0f / (float)B;
  const float mu   = s * invB;
  const float var  = s2 * invB - mu * mu;
  const float sc   = gamma[col] * rsqrtf(var + 1e-5f);
  const float be   = beta[col];
  for (int r = 0; r < B; ++r) {
    float y = sc * (logits[r * 10 + col] - mu) + be;
    logits[r * 10 + col] = y > 0.0f ? y : 0.0f;
  }
}

__global__ void logsoftmax_kernel(const float* __restrict__ logits,
                                  float* __restrict__ outp, int B) {
  const int b = blockIdx.x * blockDim.x + threadIdx.x;
  if (b >= B) return;
  float mx = -1e30f;
  for (int j = 0; j < 10; ++j) mx = fmaxf(mx, logits[b * 10 + j]);
  float s = 0.0f;
  for (int j = 0; j < 10; ++j) s += expf(logits[b * 10 + j] - mx);
  const float lse = mx + logf(s);
  for (int j = 0; j < 10; ++j) outp[b * 10 + j] = logits[b * 10 + j] - lse;
}

// ---------------------------------------------------------------------------
extern "C" void kernel_launch(void* const* d_in, const int* in_sizes, int n_in,
                              void* d_out, int out_size, void* d_ws,
                              size_t ws_size, hipStream_t stream) {
  (void)in_sizes; (void)out_size; (void)ws_size;
  static const int IN[4][5]   = {{4,8,4,8,3},{8,8,8,8,8},{8,4,8,4,8},{4,4,4,4,4}};
  static const int OUTM[4][5] = {{8,8,8,8,8},{8,4,8,4,8},{4,4,4,4,4},{4,4,4,4,4}};
  static const int R[6] = {1,4,4,4,4,1};
  const int B = 512;

  const float* x = (const float*)d_in[0];
  const float* cores[4][5];
  const float *bias[4], *gam[4], *bet[4], *msk[4];
  const float *w5, *b5, *g5, *be5;

  if (n_in >= 41) {                 // pytree-flattened cores tuples
    int idx = 1;
    for (int l = 0; l < 4; ++l) {
      for (int k = 0; k < 5; ++k) cores[l][k] = (const float*)d_in[idx++];
      bias[l] = (const float*)d_in[idx++];
      gam[l]  = (const float*)d_in[idx++];
      bet[l]  = (const float*)d_in[idx++];
      msk[l]  = (const float*)d_in[idx++];
    }
    w5 = (const float*)d_in[idx++]; b5 = (const float*)d_in[idx++];
    g5 = (const float*)d_in[idx++]; be5 = (const float*)d_in[idx++];
  } else {                          // cores tuples as one concatenated buffer
    int idx = 1;
    for (int l = 0; l < 4; ++l) {
      const float* base = (const float*)d_in[idx++];
      size_t off = 0;
      for (int k = 0; k < 5; ++k) {
        cores[l][k] = base + off;
        off += (size_t)R[k] * IN[l][k] * OUTM[l][k] * R[k + 1];
      }
      bias[l] = (const float*)d_in[idx++];
      gam[l]  = (const float*)d_in[idx++];
      bet[l]  = (const float*)d_in[idx++];
      msk[l]  = (const float*)d_in[idx++];
    }
    w5 = (const float*)d_in[idx++]; b5 = (const float*)d_in[idx++];
    g5 = (const float*)d_in[idx++]; be5 = (const float*)d_in[idx++];
  }

  // Workspace (f16 intermediates): region0 = 2^26 halves (max even-position
  // intermediate 512*4*4096*8), region1 = 2^25 halves (max odd-position),
  // then f32 logits (5120). Total ~192 MiB + 20 KiB.
  _Float16* ws = (_Float16*)d_ws;
  _Float16* region[2] = { ws, ws + ((size_t)1 << 26) };
  float* logits = (float*)(ws + ((size_t)1 << 26) + ((size_t)1 << 25));

  // 20 fully specialized step launchers, in execution order.
  static const StepFn steps[20] = {
    // layer 1: in {4,8,4,8,3} -> out {8,8,8,8,8}  (step 0 reads f32 input x)
    launch_step<float,     4,  768, 8, 4>, launch_step<_Float16, 32,  768, 8, 4>,
    launch_step<_Float16, 16, 1536, 8, 4>, launch_step<_Float16, 32, 1536, 8, 4>,
    launch_step<_Float16, 12, 4096, 8, 1>,
    // layer 2: in {8,8,8,8,8} -> out {8,4,8,4,8}
    launch_step<_Float16,  8, 4096, 8, 4>, launch_step<_Float16, 32, 4096, 4, 4>,
    launch_step<_Float16, 32, 2048, 8, 4>, launch_step<_Float16, 32, 2048, 4, 4>,
    launch_step<_Float16, 32, 1024, 8, 1>,
    // layer 3: in {8,4,8,4,8} -> out {4,4,4,4,4}
    launch_step<_Float16,  8, 1024, 4, 4>, launch_step<_Float16, 16, 1024, 4, 4>,
    launch_step<_Float16, 32,  512, 4, 4>, launch_step<_Float16, 16,  512, 4, 4>,
    launch_step<_Float16, 32,  256, 4, 1>,
    // layer 4: in {4,4,4,4,4} -> out {4,4,4,4,4}
    launch_step<_Float16,  4,  256, 4, 4>, launch_step<_Float16, 16,  256, 4, 4>,
    launch_step<_Float16, 16,  256, 4, 4>, launch_step<_Float16, 16,  256, 4, 4>,
    launch_step<_Float16, 16,  256, 4, 1>,
  };

  const void* src = x;
  int dst = 0;
  int step = 0;
  for (int l = 0; l < 4; ++l) {
    int Nl = 1;
    for (int k = 0; k < 5; ++k) Nl *= OUTM[l][k];
    for (int k = 0; k < 5; ++k) {
      steps[step](src, cores[l][k], region[dst], stream);
      src = region[dst]; dst ^= 1;
      ++step;
    }
    bn_relu_mask_kernel<<<(Nl + 255) / 256, 256, 0, stream>>>(
        (const _Float16*)src, bias[l], gam[l], bet[l], msk[l], region[dst],
        B, Nl);
    src = region[dst]; dst ^= 1;
  }

  dense10_kernel<<<(B * 10 + 255) / 256, 256, 0, stream>>>(
      (const _Float16*)src, w5, b5, logits, B);
  bn5_kernel<<<1, 32, 0, stream>>>(logits, g5, be5, B);
  logsoftmax_kernel<<<(B + 255) / 256, 256, 0, stream>>>(logits,
                                                         (float*)d_out, B);
}
